// PyTorchAttentionLayer_44375602102696
// MI455X (gfx1250) — matrix operand order repaired and checked by reading, not compiled, not run on hardware
//
#include <hip/hip_runtime.h>

// ---------------- problem constants ----------------
constexpr int cS  = 512;
constexpr int cB  = 4;
constexpr int cE  = 768;
constexpr int cH  = 12;
constexpr int cDH = 64;
constexpr int cNR = cS * cB;            // 2048 token rows
constexpr long cR = (long)cB * cS * cS; // 1,048,576 bias rows

// ---------------- WMMA types ----------------
typedef __attribute__((ext_vector_type(16))) __bf16 v16bf;
typedef __attribute__((ext_vector_type(8)))  float  v8f;

union FragBF {
    v16bf v;
    unsigned short u[16];
    unsigned int   d[8];
};

#define WMMA_BF16(A, B, C) \
    __builtin_amdgcn_wmma_f32_16x16x32_bf16(false, (A), false, (B), (short)0, (C), false, false)

// ---------------- async global->LDS path (CDNA5) ----------------
// Builtin prototype (from hipcc diagnostic): (v2i AS1*, v2i AS3*, imm, imm)
#if __has_builtin(__builtin_amdgcn_global_load_async_to_lds_b64) && \
    __has_builtin(__builtin_amdgcn_s_wait_asynccnt)
#define HAVE_ASYNC_LDS 1
typedef int async_v2i __attribute__((vector_size(8)));
__device__ __forceinline__ void async_cp_b64(const void* g, void* l) {
    __builtin_amdgcn_global_load_async_to_lds_b64(
        (__attribute__((address_space(1))) async_v2i*)(g),
        (__attribute__((address_space(3))) async_v2i*)(l), 0, 0);
}
__device__ __forceinline__ void async_wait0() {
    __builtin_amdgcn_s_wait_asynccnt(0);
}
#else
#define HAVE_ASYNC_LDS 0
#endif

__device__ __forceinline__ unsigned short f2bf(float f) {
    unsigned int u = __float_as_uint(f);
    u += 0x7FFFu + ((u >> 16) & 1u);   // round-to-nearest-even
    return (unsigned short)(u >> 16);
}

// K position held by VGPR-pair v for 16-bit A matrix (16x32), contiguous order:
// lanes 0-15: K={0..15}, lanes 16-31: K={16..31}, 2 per VGPR.
__device__ __forceinline__ int a_kpat(int v, int laneHi) {
    return 2 * v + laneHi * 16;
}

// =====================================================================
// Kernel 1: q/k/v projections  Y = X @ W^T + b   (scatter to [B,H,S,DH] bf16)
// grid (12, 16, 3)  block 256 ; block tile 128(M) x 64(N), K-step 32
// =====================================================================
__global__ __launch_bounds__(256)
void proj_kernel(const float* __restrict__ q, const float* __restrict__ k,
                 const float* __restrict__ v, const float* __restrict__ w_all,
                 const float* __restrict__ b_all,
                 unsigned short* __restrict__ qh, unsigned short* __restrict__ kh,
                 unsigned short* __restrict__ vh)
{
    const int mat = blockIdx.z;
    const float* X    = (mat == 0) ? q : (mat == 1) ? k : v;
    const float* W    = w_all + (long)mat * cE * cE;
    const float* bias = b_all + mat * cE;
    unsigned short* dst = (mat == 0) ? qh : (mat == 1) ? kh : vh;
    const float outscale = (mat == 0) ? 0.125f : 1.0f;   // 1/sqrt(DH) folded into q

    __shared__ unsigned short lA[128][36];
    __shared__ unsigned short lB[64][36];

    const int t      = threadIdx.x;
    const int lane   = t & 31;
    const int wv     = t >> 5;       // wave 0..7
    const int wm     = wv >> 1;      // 0..3
    const int wn     = wv & 1;       // 0..1
    const int laneHi = lane >> 4;
    const int ln     = lane & 15;

    const int row0 = blockIdx.y * 128;
    const int col0 = blockIdx.x * 64;

    v8f acc[2][2];
#pragma unroll
    for (int i = 0; i < 2; i++)
#pragma unroll
        for (int j = 0; j < 2; j++) acc[i][j] = (v8f){0,0,0,0,0,0,0,0};

    for (int kk = 0; kk < cE; kk += 32) {
        __syncthreads();
        // A tile 128x32 f32 -> bf16 (1024 float4, 4/thread)
#pragma unroll
        for (int it = 0; it < 4; it++) {
            int idx = t + it * 256;
            int r = idx >> 3, c4 = idx & 7;
            float4 f = *(const float4*)(X + (long)(row0 + r) * cE + kk + c4 * 4);
            unsigned short* p = &lA[r][c4 * 4];
            p[0] = f2bf(f.x); p[1] = f2bf(f.y); p[2] = f2bf(f.z); p[3] = f2bf(f.w);
        }
        // B tile 64x32 (rows = output features) (512 float4, 2/thread)
#pragma unroll
        for (int it = 0; it < 2; it++) {
            int idx = t + it * 256;
            int r = idx >> 3, c4 = idx & 7;
            float4 f = *(const float4*)(W + (long)(col0 + r) * cE + kk + c4 * 4);
            unsigned short* p = &lB[r][c4 * 4];
            p[0] = f2bf(f.x); p[1] = f2bf(f.y); p[2] = f2bf(f.z); p[3] = f2bf(f.w);
        }
        __syncthreads();

        FragBF a[2], bfr[2];
#pragma unroll
        for (int mi = 0; mi < 2; mi++) {
            int m = wm * 32 + mi * 16 + ln;
#pragma unroll
            for (int vv = 0; vv < 8; vv++) {
                int kp = a_kpat(vv, laneHi);
                a[mi].u[2 * vv]     = lA[m][kp];
                a[mi].u[2 * vv + 1] = lA[m][kp + 1];
            }
        }
#pragma unroll
        for (int ni = 0; ni < 2; ni++) {
            int n = wn * 32 + ni * 16 + ln;
#pragma unroll
            for (int vv = 0; vv < 8; vv++) {
                int kp = laneHi * 16 + 2 * vv;
                bfr[ni].u[2 * vv]     = lB[n][kp];
                bfr[ni].u[2 * vv + 1] = lB[n][kp + 1];
            }
        }
#pragma unroll
        for (int mi = 0; mi < 2; mi++)
#pragma unroll
            for (int ni = 0; ni < 2; ni++)
                acc[mi][ni] = WMMA_BF16(a[mi].v, bfr[ni].v, acc[mi][ni]);
    }

    // epilogue: scatter into [B,H,S,DH]
#pragma unroll
    for (int mi = 0; mi < 2; mi++)
#pragma unroll
        for (int ni = 0; ni < 2; ni++)
#pragma unroll
            for (int vv = 0; vv < 8; vv++) {
                int r = row0 + wm * 32 + mi * 16 + vv + laneHi * 8;
                int o = col0 + wn * 32 + ni * 16 + ln;
                float y = (acc[mi][ni][vv] + bias[o]) * outscale;
                int s = r >> 2, bb = r & 3;
                int h = o >> 6, dd = o & 63;
                dst[((long)((bb * cH + h) * cS + s)) * cDH + dd] = f2bf(y);
            }
}

// =====================================================================
// Kernel 2: edge bias  bias[b,h,i,j] = p[b,i,j,:] . edge_w[h,:] + edge_b[h]
// rows = (b,i,j) flattened: [1M x 64] @ [64 x 16pad]; grid 8192, block 256
// =====================================================================
__global__ __launch_bounds__(256)
void bias_kernel(const float* __restrict__ p, const float* __restrict__ ew,
                 const float* __restrict__ eb, float* __restrict__ biasOut)
{
    __shared__ unsigned short lP[128][72];
    __shared__ unsigned short lE[16][64];

    const int t      = threadIdx.x;
    const int lane   = t & 31;
    const int wv     = t >> 5;
    const int laneHi = lane >> 4;
    const int ln     = lane & 15;
    const long row0  = (long)blockIdx.x * 128;

    // edge_w -> LDS bf16, pad heads 12..15 with zero
#pragma unroll
    for (int it = 0; it < 4; it++) {
        int idx = t + it * 256;           // 0..1023
        int h = idx >> 6, c = idx & 63;
        float val = (h < cH) ? ew[h * cDH + c] : 0.f;
        lE[h][c] = f2bf(val);
    }
    // p tile 128x64 f32 -> bf16 (2048 float4, 8/thread) — the big 268MB stream
#pragma unroll
    for (int it = 0; it < 8; it++) {
        int idx = t + it * 256;
        int r = idx >> 4, c4 = idx & 15;
        float4 f = *(const float4*)(p + (row0 + r) * 64 + c4 * 4);
        unsigned short* q = &lP[r][c4 * 4];
        q[0] = f2bf(f.x); q[1] = f2bf(f.y); q[2] = f2bf(f.z); q[3] = f2bf(f.w);
    }
    __syncthreads();

    FragBF a0, a1, b0, b1;
    const int m = wv * 16 + ln;
#pragma unroll
    for (int vv = 0; vv < 8; vv++) {
        int kp = a_kpat(vv, laneHi);
        a0.u[2 * vv]     = lP[m][kp];
        a0.u[2 * vv + 1] = lP[m][kp + 1];
        a1.u[2 * vv]     = lP[m][kp + 32];
        a1.u[2 * vv + 1] = lP[m][kp + 33];
        int kb = laneHi * 16 + 2 * vv;
        b0.u[2 * vv]     = lE[ln][kb];
        b0.u[2 * vv + 1] = lE[ln][kb + 1];
        b1.u[2 * vv]     = lE[ln][kb + 32];
        b1.u[2 * vv + 1] = lE[ln][kb + 33];
    }
    v8f c = (v8f){0,0,0,0,0,0,0,0};
    c = WMMA_BF16(a0.v, b0.v, c);
    c = WMMA_BF16(a1.v, b1.v, c);

#pragma unroll
    for (int vv = 0; vv < 8; vv++) {
        if (ln < cH) {
            long r = row0 + wv * 16 + vv + laneHi * 8;
            int bb  = (int)(r >> 18);
            int rem = (int)(r & 262143);
            int i = rem >> 9, j = rem & 511;
            biasOut[(((long)(bb * cH + ln) * cS + i) * cS) + j] = c[vv] + eb[ln];
        }
    }
}

// =====================================================================
// Kernel 3: flash attention with precomputed additive bias.
// grid (B*H=48, S/128=4), block 256 = 8 waves; each wave owns 16 i-rows,
// loops j in steps of 32 (2 score tiles -> one K=32 P fragment).
// K/V tiles staged with GLOBAL_LOAD_ASYNC_TO_LDS (ASYNCcnt) when available.
// =====================================================================
__global__ __launch_bounds__(256)
void attn_kernel(const unsigned short* __restrict__ qh,
                 const unsigned short* __restrict__ kh,
                 const unsigned short* __restrict__ vh,
                 const float* __restrict__ biasIn,
                 unsigned short* __restrict__ ctx)
{
    __shared__ unsigned short lK[32][68];   // row stride 136 B (8B aligned, no bank conflicts)
    __shared__ unsigned short lV[32][68];
    __shared__ unsigned short lPb[8][16][40];

    const int t      = threadIdx.x;
    const int lane   = t & 31;
    const int wv     = t >> 5;
    const int laneHi = lane >> 4;
    const int ln     = lane & 15;

    const int bh = blockIdx.x;           // b*H + h
    const int i0 = blockIdx.y * 128;
    const int bb = bh / cH, h = bh % cH;

    const unsigned short* qbase = qh + (long)bh * cS * cDH;
    const unsigned short* kbase = kh + (long)bh * cS * cDH;
    const unsigned short* vbase = vh + (long)bh * cS * cDH;
    const float* bbase = biasIn + (long)bh * cS * cS;

    const int iw = i0 + wv * 16;

    // Q fragments (A, 2 K-steps over DH=64), q already scaled by 1/8
    FragBF aq[2];
    {
        const unsigned short* qrow = qbase + (long)(iw + ln) * cDH;
#pragma unroll
        for (int ks = 0; ks < 2; ks++)
#pragma unroll
            for (int vv = 0; vv < 8; vv++) {
                int kp = ks * 32 + a_kpat(vv, laneHi);
                aq[ks].d[vv] = *(const unsigned int*)(qrow + kp);
            }
    }

    v8f oacc[4];
#pragma unroll
    for (int g = 0; g < 4; g++) oacc[g] = (v8f){0,0,0,0,0,0,0,0};
    float mrow[8], lrow[8];
#pragma unroll
    for (int vv = 0; vv < 8; vv++) { mrow[vv] = -1e30f; lrow[vv] = 0.f; }

    for (int j0 = 0; j0 < cS; j0 += 32) {
        __syncthreads();
        // stage K,V tiles 32x64 bf16
#if HAVE_ASYNC_LDS
        // 512 b64 transfers per tensor, 2/thread/tensor; LDS targets 8B aligned
#pragma unroll
        for (int it = 0; it < 2; it++) {
            int idx = t + it * 256;       // 0..511
            int r = idx >> 4, cc = idx & 15;
            async_cp_b64(kbase + (long)(j0 + r) * cDH + cc * 4, &lK[r][cc * 4]);
            async_cp_b64(vbase + (long)(j0 + r) * cDH + cc * 4, &lV[r][cc * 4]);
        }
        async_wait0();
#else
#pragma unroll
        for (int it = 0; it < 2; it++) {
            int idx = t + it * 256;       // 0..511
            int r = idx >> 4, cc = idx & 15;
            *(uint2*)&lK[r][cc * 4] = *(const uint2*)(kbase + (long)(j0 + r) * cDH + cc * 4);
            *(uint2*)&lV[r][cc * 4] = *(const uint2*)(vbase + (long)(j0 + r) * cDH + cc * 4);
        }
#endif
        __syncthreads();

        // scores: C initialized with bias tile, then 2 WMMAs (K=64) per j-half
        v8f sc[2];
#pragma unroll
        for (int jh = 0; jh < 2; jh++) {
#pragma unroll
            for (int vv = 0; vv < 8; vv++)
                sc[jh][vv] = bbase[(long)(iw + vv + laneHi * 8) * cS + j0 + jh * 16 + ln];
#pragma unroll
            for (int ks = 0; ks < 2; ks++) {
                FragBF bk;
#pragma unroll
                for (int vv = 0; vv < 8; vv++) {
                    int kp = ks * 32 + laneHi * 16 + 2 * vv;
                    bk.d[vv] = *(const unsigned int*)&lK[jh * 16 + ln][kp];
                }
                sc[jh] = WMMA_BF16(aq[ks].v, bk.v, sc[jh]);
            }
        }

        // online softmax over the 32-wide j tile
#pragma unroll
        for (int vv = 0; vv < 8; vv++) {
            float tm = fmaxf(sc[0][vv], sc[1][vv]);
#pragma unroll
            for (int msk = 1; msk < 16; msk <<= 1) tm = fmaxf(tm, __shfl_xor(tm, msk, 32));
            float mn = fmaxf(mrow[vv], tm);
            float alpha = __expf(mrow[vv] - mn);
            mrow[vv] = mn;
            float p0 = __expf(sc[0][vv] - mn);
            float p1 = __expf(sc[1][vv] - mn);
            sc[0][vv] = p0; sc[1][vv] = p1;
            float ts = p0 + p1;
#pragma unroll
            for (int msk = 1; msk < 16; msk <<= 1) ts += __shfl_xor(ts, msk, 32);
            lrow[vv] = lrow[vv] * alpha + ts;
#pragma unroll
            for (int g = 0; g < 4; g++) oacc[g][vv] *= alpha;
        }

        // transpose P (C layout -> A layout) through per-wave LDS
#pragma unroll
        for (int jh = 0; jh < 2; jh++)
#pragma unroll
            for (int vv = 0; vv < 8; vv++)
                lPb[wv][vv + laneHi * 8][jh * 16 + ln] = f2bf(sc[jh][vv]);
        __syncthreads();

        FragBF ap;
#pragma unroll
        for (int vv = 0; vv < 8; vv++) {
            int kp = a_kpat(vv, laneHi);
            ap.u[2 * vv]     = lPb[wv][ln][kp];
            ap.u[2 * vv + 1] = lPb[wv][ln][kp + 1];
        }
        // context update: O += P(16x32) @ V(32x64), 4 d-groups
#pragma unroll
        for (int g = 0; g < 4; g++) {
            FragBF bv;
#pragma unroll
            for (int vv = 0; vv < 8; vv++) {
                int jl = laneHi * 16 + 2 * vv;
                bv.u[2 * vv]     = lV[jl][g * 16 + ln];
                bv.u[2 * vv + 1] = lV[jl + 1][g * 16 + ln];
            }
            oacc[g] = WMMA_BF16(ap.v, bv.v, oacc[g]);
        }
    }

    // normalize and write ctx in [S,B,E] bf16 layout (row = s*B+b, col = h*DH+d)
#pragma unroll
    for (int g = 0; g < 4; g++)
#pragma unroll
        for (int vv = 0; vv < 8; vv++) {
            int i = iw + vv + laneHi * 8;
            int d = h * cDH + g * 16 + ln;
            float val = oacc[g][vv] / lrow[vv];
            ctx[(long)(i * cB + bb) * cE + d] = f2bf(val);
        }
}

// =====================================================================
// Kernel 4: out = ctx @ out_proj_w^T + out_proj_b  -> f32 [S,B,E]
// grid (12, 16), block 256; tile 128x64, K-step 32; A already bf16 so the
// A tile is staged with GLOBAL_LOAD_ASYNC_TO_LDS when available.
// =====================================================================
__global__ __launch_bounds__(256)
void outproj_kernel(const unsigned short* __restrict__ A,
                    const float* __restrict__ W,
                    const float* __restrict__ bias,
                    float* __restrict__ out)
{
    __shared__ unsigned short lA[128][36];  // row stride 72 B (8B aligned)
    __shared__ unsigned short lB[64][36];

    const int t      = threadIdx.x;
    const int lane   = t & 31;
    const int wv     = t >> 5;
    const int wm     = wv >> 1;
    const int wn     = wv & 1;
    const int laneHi = lane >> 4;
    const int ln     = lane & 15;

    const int row0 = blockIdx.y * 128;
    const int col0 = blockIdx.x * 64;

    v8f acc[2][2];
#pragma unroll
    for (int i = 0; i < 2; i++)
#pragma unroll
        for (int j = 0; j < 2; j++) acc[i][j] = (v8f){0,0,0,0,0,0,0,0};

    for (int kk = 0; kk < cE; kk += 32) {
        __syncthreads();
        // A tile 128x32 bf16
#if HAVE_ASYNC_LDS
        // 1024 b64 transfers, 4/thread; LDS targets 8B aligned
#pragma unroll
        for (int it = 0; it < 4; it++) {
            int idx = t + it * 256;       // 0..1023
            int r = idx >> 3, cc = idx & 7;
            async_cp_b64(A + (long)(row0 + r) * cE + kk + cc * 4, &lA[r][cc * 4]);
        }
#else
#pragma unroll
        for (int it = 0; it < 2; it++) {
            int idx = t + it * 256;
            int r = idx >> 2, cc = idx & 3;
            uint4 f = *(const uint4*)(A + (long)(row0 + r) * cE + kk + cc * 8);
            unsigned int* p = (unsigned int*)&lA[r][cc * 8];
            p[0] = f.x; p[1] = f.y; p[2] = f.z; p[3] = f.w;
        }
#endif
        // W tile 64x32 f32 -> bf16 (needs VALU convert, stays synchronous)
#pragma unroll
        for (int it = 0; it < 2; it++) {
            int idx = t + it * 256;
            int r = idx >> 3, c4 = idx & 7;
            float4 f = *(const float4*)(W + (long)(col0 + r) * cE + kk + c4 * 4);
            unsigned short* p = &lB[r][c4 * 4];
            p[0] = f2bf(f.x); p[1] = f2bf(f.y); p[2] = f2bf(f.z); p[3] = f2bf(f.w);
        }
#if HAVE_ASYNC_LDS
        async_wait0();
#endif
        __syncthreads();

        FragBF a[2], bfr[2];
#pragma unroll
        for (int mi = 0; mi < 2; mi++) {
            int m = wm * 32 + mi * 16 + ln;
#pragma unroll
            for (int vv = 0; vv < 8; vv++) {
                int kp = a_kpat(vv, laneHi);
                a[mi].u[2 * vv]     = lA[m][kp];
                a[mi].u[2 * vv + 1] = lA[m][kp + 1];
            }
        }
#pragma unroll
        for (int ni = 0; ni < 2; ni++) {
            int n = wn * 32 + ni * 16 + ln;
#pragma unroll
            for (int vv = 0; vv < 8; vv++) {
                int kp = laneHi * 16 + 2 * vv;
                bfr[ni].u[2 * vv]     = lB[n][kp];
                bfr[ni].u[2 * vv + 1] = lB[n][kp + 1];
            }
        }
#pragma unroll
        for (int mi = 0; mi < 2; mi++)
#pragma unroll
            for (int ni = 0; ni < 2; ni++)
                acc[mi][ni] = WMMA_BF16(a[mi].v, bfr[ni].v, acc[mi][ni]);
    }

#pragma unroll
    for (int mi = 0; mi < 2; mi++)
#pragma unroll
        for (int ni = 0; ni < 2; ni++)
#pragma unroll
            for (int vv = 0; vv < 8; vv++) {
                int r = row0 + wm * 32 + mi * 16 + vv + laneHi * 8;
                int o = col0 + wn * 32 + ni * 16 + ln;
                out[(long)r * cE + o] = acc[mi][ni][vv] + bias[o];
            }
}

// =====================================================================
// launch
// =====================================================================
extern "C" void kernel_launch(void* const* d_in, const int* in_sizes, int n_in,
                              void* d_out, int out_size, void* d_ws, size_t ws_size,
                              hipStream_t stream) {
    const float* q   = (const float*)d_in[0];
    const float* k   = (const float*)d_in[1];
    const float* v   = (const float*)d_in[2];
    const float* p   = (const float*)d_in[3];
    const float* ipw = (const float*)d_in[4];
    const float* ipb = (const float*)d_in[5];
    const float* opw = (const float*)d_in[6];
    const float* opb = (const float*)d_in[7];
    const float* ew  = (const float*)d_in[8];
    const float* eb  = (const float*)d_in[9];

    // workspace layout (needs ~63 MB):
    //   qh/kh/vh/ctx : 4 x 1,572,864 bf16  = 12,582,912 B
    //   bias         : 12,582,912 f32      = 50,331,648 B
    const long NHD = (long)cB * cH * cS * cDH;   // 1,572,864
    unsigned short* qh  = (unsigned short*)d_ws;
    unsigned short* kh  = qh + NHD;
    unsigned short* vh  = kh + NHD;
    unsigned short* ctx = vh + NHD;
    float* biasbuf = (float*)(ctx + NHD);

    proj_kernel<<<dim3(cE / 64, cNR / 128, 3), 256, 0, stream>>>(
        q, k, v, ipw, ipb, qh, kh, vh);

    bias_kernel<<<dim3((unsigned)(cR / 128)), 256, 0, stream>>>(
        p, ew, eb, biasbuf);

    attn_kernel<<<dim3(cB * cH, cS / 128), 256, 0, stream>>>(
        qh, kh, vh, biasbuf, ctx);

    outproj_kernel<<<dim3(cE / 64, cNR / 128), 256, 0, stream>>>(
        ctx, opw, opb, (float*)d_out);
}
